// ConvPointnet_Encoder_89481348644934
// MI455X (gfx1250) — matrix operand order repaired and checked by reading, not compiled, not run on hardware
//
#include <hip/hip_runtime.h>

// ---------------------------------------------------------------------------
// CDNA5 (gfx1250) WMMA implementation of the ConvPointnet encoder.
//   - all dense math on v_wmma_f32_16x16x32_f16 (f32 accumulate)
//   - point-MLP GEMMs: weight panel (128 x K<=256 f16 = 64KB) staged into LDS
//     by the Tensor Data Mover (tensor_load_to_lds + s_wait_tensorcnt) when
//     available, else a cooperative b128 copy; 8-wave blocks, each wave =
//     16 rows x all 8 N-tiles -> 8 WMMAs per A-fragment load
//   - UNet convs: implicit-GEMM, NHWC f16, b128 fragment loads, border
//     handling via address clamp + v_cndmask (no divergent branches)
//   - relu fused into GEMM stores where the consumer allows it
// ---------------------------------------------------------------------------

typedef _Float16 h16;
typedef __attribute__((ext_vector_type(16))) _Float16 v16h;
typedef __attribute__((ext_vector_type(8)))  _Float16 v8h;
typedef __attribute__((ext_vector_type(8)))  float    v8f;
typedef __attribute__((ext_vector_type(4)))  unsigned u32x4;
typedef __attribute__((ext_vector_type(8)))  int      i32x8;
typedef __attribute__((ext_vector_type(4)))  int      i32x4;

#if __has_builtin(__builtin_amdgcn_tensor_load_to_lds) && \
    __has_builtin(__builtin_amdgcn_s_wait_tensorcnt)
#define USE_TDM 1
#else
#define USE_TDM 0
#endif

#define RR      64
#define RSQ     4096
#define NBATCH  4
#define NPTS    30000
#define MPTS    (NBATCH * NPTS)
#define HIDD    128
#define CDIMC   128
#define LATC    32
#define PADCF   0.1f

// Monotonic float <-> uint mapping so atomicMax(uint) implements float max.
__device__ __forceinline__ unsigned fkey(float f) {
  unsigned u = __float_as_uint(f);
  return (u & 0x80000000u) ? ~u : (u | 0x80000000u);
}
__device__ __forceinline__ float fdec(unsigned u) {
  return __uint_as_float((u & 0x80000000u) ? (u ^ 0x80000000u) : ~u);
}

// two 16-byte (8 x f16) loads -> one 16-element WMMA fragment
__device__ __forceinline__ v16h ld_frag(const h16* p0, const h16* p1) {
  v8h lo = *(const v8h*)p0;
  v8h hi = *(const v8h*)p1;
  return __builtin_shufflevector(lo, hi, 0, 1, 2, 3, 4, 5, 6, 7,
                                 8, 9, 10, 11, 12, 13, 14, 15);
}

__device__ __forceinline__ v16h relu16(v16h a) {
#pragma unroll
  for (int e = 0; e < 16; ++e) a[e] = __builtin_fmaxf16(a[e], (h16)0.f);
  return a;
}

// ---------------------------------------------------------------------------
// Fills
// ---------------------------------------------------------------------------
__global__ void fill_f32_kernel(float* p, float v, int n) {
  int i = blockIdx.x * blockDim.x + threadIdx.x;
  if (i < n) p[i] = v;
}
__global__ void fill_u32_kernel(unsigned* p, unsigned v, int n) {
  int i = blockIdx.x * blockDim.x + threadIdx.x;
  if (i < n) p[i] = v;
}

// ---------------------------------------------------------------------------
// Weight conversions (once per call)
// FC:   W[K,N] f32 -> WT[N][K] f16      (B-fragment reads 16 contiguous K)
// Conv: W OIHW f32 -> WT[co][tap][ci] f16
// ---------------------------------------------------------------------------
__global__ void cvt_fcw_kernel(const float* __restrict__ W, h16* __restrict__ WT,
                               int K, int N) {
  int i = blockIdx.x * blockDim.x + threadIdx.x;
  if (i >= K * N) return;
  int n = i / K, k = i % K;
  WT[i] = (h16)W[(size_t)k * N + n];
}
__global__ void cvt_convw_kernel(const float* __restrict__ W, h16* __restrict__ WT,
                                 int Cout, int Cin, int ks2) {
  int i = blockIdx.x * blockDim.x + threadIdx.x;
  if (i >= Cout * ks2 * Cin) return;
  int ci = i % Cin, t = (i / Cin) % ks2, co = i / (Cin * ks2);
  WT[i] = (h16)W[((size_t)co * Cin + ci) * ks2 + t];
}

// ---------------------------------------------------------------------------
// fc_pos: [M,3] @ [3,256] + b -> f16  (K=3, not worth WMMA)
// ---------------------------------------------------------------------------
__global__ void fc_pos_kernel(const float* __restrict__ emb,
                              const float* __restrict__ w,
                              const float* __restrict__ b,
                              h16* __restrict__ out) {
  int i = blockIdx.x * blockDim.x + threadIdx.x;
  if (i >= MPTS * 256) return;
  int r = i >> 8, o = i & 255;
  float acc = b[o];
#pragma unroll
  for (int j = 0; j < 3; ++j) acc += emb[r * 3 + j] * w[j * 256 + o];
  out[i] = (h16)acc;
}

// ---------------------------------------------------------------------------
// Plane segment ids; planes: (x,z), (x,y), (y,z)
// ---------------------------------------------------------------------------
__global__ void ids_kernel(const float* __restrict__ p, int* __restrict__ ids) {
  int i = blockIdx.x * blockDim.x + threadIdx.x;
  if (i >= MPTS) return;
  int b = i / NPTS;
  float cc[3] = {p[i * 3 + 0], p[i * 3 + 1], p[i * 3 + 2]};
  const int a0[3] = {0, 0, 1};
  const int a1[3] = {2, 1, 2};
  const float inv = 1.0f / (2.0f * (1.0f + PADCF + 1e-5f));
#pragma unroll
  for (int k = 0; k < 3; ++k) {
    float u = cc[a0[k]] * inv + 0.5f;
    float v = cc[a1[k]] * inv + 0.5f;
    u = fminf(fmaxf(u, 0.0f), 1.0f - 1e-5f);
    v = fminf(fmaxf(v, 0.0f), 1.0f - 1e-5f);
    ids[k * MPTS + i] = (int)(u * RR) + RR * (int)(v * RR) + b * RSQ;
  }
}

// ---------------------------------------------------------------------------
// Scatter-max of net[M,128] (f16) into 3 plane grids (uint-keyed float max)
// ---------------------------------------------------------------------------
__global__ void scatter_max_kernel(const h16* __restrict__ net,
                                   const int* __restrict__ ids,
                                   unsigned* __restrict__ grid) {
  int i = blockIdx.x * blockDim.x + threadIdx.x;
  if (i >= MPTS * HIDD) return;
  int r = i >> 7, f = i & 127;
  unsigned key = fkey((float)net[i]);
#pragma unroll
  for (int k = 0; k < 3; ++k) {
    int seg = ids[k * MPTS + r];
    atomicMax(&grid[((size_t)k * NBATCH * RSQ + seg) * HIDD + f], key);
  }
}

// Gather pooled (sum over 3 planes) and build concat [net | pooled] f16
__global__ void gather_concat_kernel(const h16* __restrict__ net,
                                     const int* __restrict__ ids,
                                     const unsigned* __restrict__ grid,
                                     h16* __restrict__ cat) {
  int i = blockIdx.x * blockDim.x + threadIdx.x;
  if (i >= MPTS * HIDD) return;
  int r = i >> 7, f = i & 127;
  float pooled = 0.0f;
#pragma unroll
  for (int k = 0; k < 3; ++k) {
    int seg = ids[k * MPTS + r];
    pooled += fdec(grid[((size_t)k * NBATCH * RSQ + seg) * HIDD + f]);
  }
  cat[(size_t)r * 256 + f] = net[i];
  cat[(size_t)r * 256 + 128 + f] = (h16)pooled;
}

// ---------------------------------------------------------------------------
// WMMA GEMM with LDS-staged weights, specialized for N = 128.
// D[M,128] = act(A)[M,K] @ W[K,128] + bias + Cadd, optional relu on store.
// Block = 256 threads (8 waves). The full WT[128][K] f16 panel (<=64KB) is
// staged into LDS -- by the Tensor Data Mover when the builtin exists
// (descriptor built per CDNA5 ISA), else by a
// cooperative b128 copy. Wave w computes rows [mt*16,+16) x all 8 N-tiles:
// one A-fragment load feeds 8 consecutive WMMAs. M%16==0, K%32==0 required.
// Fragment layouts:
//  A (16-bit 16x32): lane L (row L&15) holds K = 16*(L>=16)+{0..15} (order used here),
//  B: lane L (col L&15) holds K = 16*(L>=16)+{0..15}.
//  C/D: acc[r] at (row r + 8*(L>=16), col L&15).
// ---------------------------------------------------------------------------
template <int K, int RELUIN, int RELUOUT, int OUTF32>
__global__ void __launch_bounds__(256)
gemm128_wmma_kernel(const h16* __restrict__ A, const h16* __restrict__ WT,
                    const float* __restrict__ bias, const float* __restrict__ Cadd,
                    void* __restrict__ Dout, int M) {
  constexpr int N = 128;
  __shared__ h16 lw[N * K];

#if USE_TDM
  if (threadIdx.x < 32) {
    // D# group 0: count=1, lds_addr, global_addr[56:0], type=2 ("image")
    unsigned lds_off = (unsigned)(size_t)(void*)lw;      // generic low 32 = LDS offset
    unsigned long long ga = (unsigned long long)(size_t)WT;
    unsigned ndw = (unsigned)(N * K / 2);                // panel size in dwords
    u32x4 g0;
    g0.x = 1u;                                           // count=1 (user descriptor)
    g0.y = lds_off;                                      // lds_addr  [63:32]
    g0.z = (unsigned)(ga & 0xffffffffu);                 // gaddr lo  [95:64]
    g0.w = (unsigned)((ga >> 32) & 0x01ffffffu) | (2u << 30); // gaddr hi + type=2
    // D# group 1: data_size=4B; 2D tensor = 1 row of ndw dwords; tile = whole row
    i32x8 g1;
    g1[0] = (2 << 16);                                   // data_size = 4 bytes
    g1[1] = (int)((ndw & 0xffffu) << 16);                // tensor_dim0[15:0] @ [63:48]
    g1[2] = (int)((ndw >> 16) & 0xffffu) | (1 << 16);    // tensor_dim0[31:16], tensor_dim1=1
    g1[3] = (int)((ndw & 0xffffu) << 16);                // tile_dim0 = ndw @ [127:112]
    g1[4] = 1;                                           // tile_dim1 = 1, tile_dim2 = 0
    g1[5] = (int)ndw;                                    // tensor_dim0_stride lo32
    g1[6] = 0;                                           // stride hi16 / dim1_stride lo16
    g1[7] = 0;
    i32x4 zz4 = {0, 0, 0, 0};                            // groups 2/3 unused (<=2D)
    i32x8 zz8 = {0, 0, 0, 0, 0, 0, 0, 0};                // extra group (unused)
    __builtin_amdgcn_tensor_load_to_lds(g0, g1, zz4, zz4, zz8, 0);
    __builtin_amdgcn_s_wait_tensorcnt(0);
  }
#else
  {
    const v8h* gsrc = (const v8h*)WT;
    v8h* ldst = (v8h*)lw;
    const int chunks = N * K / 8;
    for (int idx = threadIdx.x; idx < chunks; idx += 256) ldst[idx] = gsrc[idx];
  }
#endif
  __syncthreads();

  int lane = threadIdx.x & 31;
  int mt = blockIdx.x * 8 + (threadIdx.x >> 5);
  if (mt >= (M >> 4)) return;
  int row0 = mt << 4;
  int hi = lane >> 4, lm = lane & 15;
  int kbA = hi << 3;   // 0 / 8
  int kbB = hi << 4;   // 0 / 16
  const h16* arow = A + (size_t)(row0 + lm) * K;

  v8f acc[8];
#pragma unroll
  for (int nt = 0; nt < 8; ++nt) {
    if (Cadd) {
#pragma unroll
      for (int rr = 0; rr < 8; ++rr)
        acc[nt][rr] = Cadd[(size_t)(row0 + rr + (hi ? 8 : 0)) * N + nt * 16 + lm];
    } else {
      acc[nt] = (v8f){};
    }
  }

#pragma unroll
  for (int k0 = 0; k0 < K; k0 += 32) {
    v16h a = ld_frag(arow + k0 + kbB, arow + k0 + kbB + 8);
    if (RELUIN) a = relu16(a);
#pragma unroll
    for (int nt = 0; nt < 8; ++nt) {
      const h16* brow = lw + (size_t)(nt * 16 + lm) * K + k0 + kbB;
      v16h bf = ld_frag(brow, brow + 8);          // ds_load_b128 x2
      acc[nt] = __builtin_amdgcn_wmma_f32_16x16x32_f16(false, a, false, bf,
                                                       (short)0, acc[nt], false, false);
    }
  }

#pragma unroll
  for (int nt = 0; nt < 8; ++nt) {
    int ncol = nt * 16 + lm;
    float bv = bias ? bias[ncol] : 0.0f;
#pragma unroll
    for (int rr = 0; rr < 8; ++rr) {
      float v = acc[nt][rr] + bv;
      if (RELUOUT) v = fmaxf(v, 0.0f);
      size_t oidx = (size_t)(row0 + rr + (hi ? 8 : 0)) * N + ncol;
      if (OUTF32) ((float*)Dout)[oidx] = v;
      else        ((h16*)Dout)[oidx] = (h16)v;
    }
  }
}

// ---------------------------------------------------------------------------
// WMMA implicit-GEMM conv (KS=1 or 3, SAME pad), NHWC f16 activations,
// WT[co][tap][ci] f16 weights. M = 16 pixels, N = 16 couts, K = 32 cins.
// Border handling: clamp address (always valid), mask fragment via cndmask.
// Requires (Bn*H*W)%16==0, Cout%16==0, Cin%32==0 (true for all calls).
// ---------------------------------------------------------------------------
template <int KS, int RELU, int OUTF32>
__global__ void __launch_bounds__(256)
conv_wmma_kernel(const h16* __restrict__ X, const h16* __restrict__ WT,
                 const float* __restrict__ bias, void* __restrict__ Yout,
                 int Bn, int Cin, int Cout, int H, int Wd) {
  constexpr int KS2 = KS * KS;
  constexpr int PAD = KS / 2;
  int wave = blockIdx.x * (blockDim.x >> 5) + (threadIdx.x >> 5);
  int lane = threadIdx.x & 31;
  int HW = H * Wd;
  int P = Bn * HW;
  int tilesN = Cout >> 4;
  if (wave >= (P >> 4) * tilesN) return;
  int tm = wave / tilesN, tn = wave % tilesN;
  int hi = lane >> 4, lm = lane & 15;
  int p = (tm << 4) + lm;
  int b = p / HW, rem = p % HW, y = rem / Wd, x = rem % Wd;
  int co = (tn << 4) + lm;
  int kbA = hi << 3, kbB = hi << 4;
  const h16* wco = WT + (size_t)co * KS2 * Cin;
  const v16h vzero = {};

  v8f acc = {};
#pragma unroll
  for (int t = 0; t < KS2; ++t) {
    const int ky = t / KS, kx = t - ky * KS;
    int yy = y + ky - PAD, xx = x + kx - PAD;
    int inb = (yy >= 0) & (yy < H) & (xx >= 0) & (xx < Wd);
    int yc = min(max(yy, 0), H - 1);
    int xc = min(max(xx, 0), Wd - 1);
    const h16* src = X + ((size_t)(b * H + yc) * Wd + xc) * Cin;
    const h16* wt = wco + (size_t)t * Cin;
    for (int k0 = 0; k0 < Cin; k0 += 32) {
      v16h a = ld_frag(src + k0 + kbB, src + k0 + kbB + 8);
      a = inb ? a : vzero;                        // v_cndmask, no branch
      v16h bf = ld_frag(wt + k0 + kbB, wt + k0 + kbB + 8);
      acc = __builtin_amdgcn_wmma_f32_16x16x32_f16(false, a, false, bf,
                                                   (short)0, acc, false, false);
    }
  }

  float bv = bias[co];
#pragma unroll
  for (int rr = 0; rr < 8; ++rr) {
    float v = acc[rr] + bv;
    if (RELU) v = fmaxf(v, 0.0f);
    size_t oidx = (size_t)((tm << 4) + rr + (hi ? 8 : 0)) * Cout + co;
    if (OUTF32) ((float*)Yout)[oidx] = v;
    else        ((h16*)Yout)[oidx] = (h16)v;
  }
}

// ---------------------------------------------------------------------------
// 2x2 maxpool stride 2, NHWC f16
// ---------------------------------------------------------------------------
__global__ void maxpool_kernel(const h16* __restrict__ X, h16* __restrict__ Y,
                               int Bn, int C, int H, int Wd) {
  int Ho = H >> 1, Wo = Wd >> 1;
  int n = Bn * Ho * Wo * C;
  int i = blockIdx.x * blockDim.x + threadIdx.x;
  if (i >= n) return;
  int c = i % C, x = (i / C) % Wo, y = (i / (C * Wo)) % Ho, b = i / (C * Wo * Ho);
  const h16* s = X + ((size_t)(b * H + 2 * y) * Wd + 2 * x) * C + c;
  float v = fmaxf(fmaxf((float)s[0], (float)s[C]),
                  fmaxf((float)s[(size_t)Wd * C], (float)s[(size_t)Wd * C + C]));
  Y[i] = (h16)v;
}

// ---------------------------------------------------------------------------
// 2x2 stride-2 transposed conv (IOHW f32 weights), NHWC f16. Writes channels
// [0,CoutH) of an output buffer with Ctot channels (for later concat).
// ---------------------------------------------------------------------------
__global__ void upconv_kernel(const h16* __restrict__ X, const float* __restrict__ Wu,
                              const float* __restrict__ bu, h16* __restrict__ Y,
                              int Bn, int Cin, int CoutH, int Hi, int Wi, int Ctot) {
  int Ho = Hi * 2, Wo = Wi * 2;
  int n = Bn * Ho * Wo * CoutH;
  int i = blockIdx.x * blockDim.x + threadIdx.x;
  if (i >= n) return;
  int co = i % CoutH, ox = (i / CoutH) % Wo, oy = (i / (CoutH * Wo)) % Ho,
      b = i / (CoutH * Wo * Ho);
  int ix = ox >> 1, iy = oy >> 1, dj = ox & 1, di = oy & 1;
  float acc = bu[co];
  const h16* xs = X + ((size_t)(b * Hi + iy) * Wi + ix) * Cin;
  const float* ws = Wu + (size_t)co * 4 + di * 2 + dj;
  for (int ci = 0; ci < Cin; ++ci)
    acc += (float)xs[ci] * ws[(size_t)ci * CoutH * 4];
  Y[((size_t)(b * Ho + oy) * Wo + ox) * Ctot + co] = (h16)acc;
}

// Copy skip feature (NHWC, Cs channels) into channels [off,off+Cs) of a
// Ctot-channel NHWC buffer.
__global__ void copy_ch_kernel(const h16* __restrict__ S, h16* __restrict__ D,
                               int nPix, int Cs, int off, int Ctot) {
  int n = nPix * Cs;
  int i = blockIdx.x * blockDim.x + threadIdx.x;
  if (i >= n) return;
  int c = i % Cs, pix = i / Cs;
  D[(size_t)pix * Ctot + off + c] = S[i];
}

// ---------------------------------------------------------------------------
// Scatter-mean accumulation of c[M,128] (f16) into 3 plane grids (f32)
// ---------------------------------------------------------------------------
__global__ void scatter_sum_kernel(const h16* __restrict__ c,
                                   const int* __restrict__ ids,
                                   float* __restrict__ gsum,
                                   float* __restrict__ gcnt) {
  int i = blockIdx.x * blockDim.x + threadIdx.x;
  if (i >= MPTS * CDIMC) return;
  int r = i >> 7, f = i & 127;
  float v = (float)c[i];
#pragma unroll
  for (int k = 0; k < 3; ++k) {
    int seg = ids[k * MPTS + r];
    atomicAdd(&gsum[((size_t)k * NBATCH * RSQ + seg) * CDIMC + f], v);
    if (f == 0) atomicAdd(&gcnt[k * NBATCH * RSQ + seg], 1.0f);
  }
}

// Finalize: CAT NHWC f16 [B,192,64,128]; plane k occupies rows [k*64,(k+1)*64)
__global__ void plane_cat_kernel(const float* __restrict__ gsum,
                                 const float* __restrict__ gcnt,
                                 h16* __restrict__ cat) {
  const int n = NBATCH * 3 * RR * RR * CDIMC;
  int i = blockIdx.x * blockDim.x + threadIdx.x;
  if (i >= n) return;
  int f = i % CDIMC;
  int w = (i / CDIMC) % RR;
  int h = (i / (CDIMC * RR)) % (3 * RR);
  int b = i / (CDIMC * RR * 3 * RR);
  int k = h / RR, yi = h % RR;
  int cell = k * NBATCH * RSQ + b * RSQ + yi * RR + w;
  float s = gsum[(size_t)cell * CDIMC + f];
  cat[i] = (h16)(s / fmaxf(gcnt[cell], 1.0f));
}

// ---------------------------------------------------------------------------
// VAE head. meanb/logvb are NHWC f32 [B,192,64,32]; outputs NCHW:
// out = [kl(4) | x(N) | mean(N) | logvar(N)], N = 4*32*192*64
// ---------------------------------------------------------------------------
__global__ void head_kernel(const float* __restrict__ meanb,
                            const float* __restrict__ logvb,
                            const float* __restrict__ noise,
                            float* __restrict__ out) {
  const int NELT = NBATCH * LATC * 3 * RSQ;
  int j = blockIdx.x * blockDim.x + threadIdx.x;
  if (j >= NELT) return;
  int w = j % RR;
  int h = (j / RR) % (3 * RR);
  int cl = (j / (RR * 3 * RR)) % LATC;
  int b = j / (RR * 3 * RR * LATC);
  int src = ((b * 3 * RR + h) * RR + w) * LATC + cl;
  float m = meanb[src];
  float lv = fminf(fmaxf(logvb[src], -30.0f), 20.0f);
  float sd = expf(0.5f * lv);
  float var = expf(lv);
  out[4 + j] = m + sd * noise[j];
  out[4 + NELT + j] = m;
  out[4 + 2 * NELT + j] = lv;
  float kl = 0.5f * (m * m + var - 1.0f - lv) * (1.0f / (float)(LATC * 3 * RSQ));
  atomicAdd(&out[b], kl);
}

// ---------------------------------------------------------------------------
// Host driver
// ---------------------------------------------------------------------------
extern "C" void kernel_launch(void* const* d_in, const int* in_sizes, int n_in,
                              void* d_out, int out_size, void* d_ws, size_t ws_size,
                              hipStream_t stream) {
  (void)in_sizes; (void)n_in; (void)out_size; (void)ws_size;

  // ---- inputs (setup_inputs dict order; params flattened in insertion order)
  int ii = 0;
  const float* p_pts = (const float*)d_in[ii++];
  const float* emb   = (const float*)d_in[ii++];
  const float* noise = (const float*)d_in[ii++];
  const float* fcpos_w = (const float*)d_in[ii++];
  const float* fcpos_b = (const float*)d_in[ii++];
  struct Blk { const float *w0, *b0, *w1, *b1, *ws; } blk[5];
  for (int k = 0; k < 5; ++k) {
    blk[k].w0 = (const float*)d_in[ii++];
    blk[k].b0 = (const float*)d_in[ii++];
    blk[k].w1 = (const float*)d_in[ii++];
    blk[k].b1 = (const float*)d_in[ii++];
    blk[k].ws = (const float*)d_in[ii++];
  }
  const float* fcc_w = (const float*)d_in[ii++];
  const float* fcc_b = (const float*)d_in[ii++];
  const float *dw[4][2], *db[4][2];
  for (int d = 0; d < 4; ++d)
    for (int c = 0; c < 2; ++c) {
      dw[d][c] = (const float*)d_in[ii++];
      db[d][c] = (const float*)d_in[ii++];
    }
  const float *uwu[3], *ubu[3], *uw[3][2], *ub[3][2];
  for (int u = 0; u < 3; ++u) {
    uwu[u]   = (const float*)d_in[ii++];
    ubu[u]   = (const float*)d_in[ii++];
    uw[u][0] = (const float*)d_in[ii++];
    ub[u][0] = (const float*)d_in[ii++];
    uw[u][1] = (const float*)d_in[ii++];
    ub[u][1] = (const float*)d_in[ii++];
  }
  const float* fin_w = (const float*)d_in[ii++];
  const float* fin_b = (const float*)d_in[ii++];
  const float* mw = (const float*)d_in[ii++];
  const float* mb = (const float*)d_in[ii++];
  const float* lw = (const float*)d_in[ii++];
  const float* lb = (const float*)d_in[ii++];
  float* outp = (float*)d_out;

  // ---- workspace carve-out (256B aligned regions)
  char* wsb = (char*)d_ws;
  size_t off = 0;
  auto carveB = [&](size_t bytes) -> char* {
    char* r = wsb + off;
    off += ((bytes + 255) & ~(size_t)255);
    return r;
  };
  auto carveH = [&](size_t n) -> h16*      { return (h16*)carveB(n * 2); };
  auto carveF = [&](size_t n) -> float*    { return (float*)carveB(n * 4); };
  auto carveI = [&](size_t n) -> int*      { return (int*)carveB(n * 4); };
  auto carveU = [&](size_t n) -> unsigned* { return (unsigned*)carveB(n * 4); };

  h16*      ACT256h = carveH((size_t)MPTS * 256);
  h16*      T1h     = carveH((size_t)MPTS * 128);
  float*    T2f     = carveF((size_t)MPTS * 128);
  h16*      ACT128h = carveH((size_t)MPTS * 128);
  h16*      C128h   = carveH((size_t)MPTS * 128);
  int*      IDS     = carveI((size_t)3 * MPTS);
  unsigned* GMAX    = carveU((size_t)3 * NBATCH * RSQ * HIDD);
  float*    GSUM    = carveF((size_t)3 * NBATCH * RSQ * CDIMC);
  float*    GCNT    = carveF((size_t)3 * NBATCH * RSQ);
  h16*      CATh    = carveH((size_t)NBATCH * 3 * RSQ * CDIMC);
  h16*      GAh     = carveH((size_t)NBATCH * 3 * RSQ * CDIMC);
  h16*      GBh     = carveH((size_t)NBATCH * 3 * RSQ * CDIMC);
  h16*      SK0h    = carveH((size_t)NBATCH * 192 * 64 * 32);
  h16*      SK1h    = carveH((size_t)NBATCH * 96 * 32 * 64);
  h16*      SK2h    = carveH((size_t)NBATCH * 48 * 16 * 128);
  float*    MEANB   = carveF((size_t)NBATCH * 3 * RSQ * LATC);
  float*    LOGVB   = carveF((size_t)NBATCH * 3 * RSQ * LATC);
  // f16 weight staging
  h16 *w0t[5], *w1t[5], *wst[5];
  for (int k = 0; k < 5; ++k) {
    w0t[k] = carveH(128 * 256);
    w1t[k] = carveH(128 * 128);
    wst[k] = carveH(128 * 256);
  }
  h16* fcct = carveH(128 * 128);
  h16 *dwt[4][2], *uwt[3][2], *fint, *mwt, *lwt;
  const int dcin[4][2]  = {{128, 32}, {32, 64}, {64, 128}, {128, 256}};
  const int dcout[4]    = {32, 64, 128, 256};
  for (int d = 0; d < 4; ++d) {
    dwt[d][0] = carveH((size_t)dcout[d] * 9 * dcin[d][0]);
    dwt[d][1] = carveH((size_t)dcout[d] * 9 * dcin[d][1]);
  }
  const int ucout[3] = {128, 64, 32};
  for (int u = 0; u < 3; ++u) {
    uwt[u][0] = carveH((size_t)ucout[u] * 9 * (2 * ucout[u]));
    uwt[u][1] = carveH((size_t)ucout[u] * 9 * ucout[u]);
  }
  fint = carveH(128 * 32);
  mwt  = carveH(32 * 128);
  lwt  = carveH(32 * 128);

  auto grid1d = [](int n) { return (n + 255) / 256; };
  auto cvt_fc = [&](const float* W, h16* WT, int K, int N) {
    cvt_fcw_kernel<<<grid1d(K * N), 256, 0, stream>>>(W, WT, K, N);
  };
  auto cvt_cv = [&](const float* W, h16* WT, int Cout, int Cin, int ks) {
    cvt_convw_kernel<<<grid1d(Cout * ks * ks * Cin), 256, 0, stream>>>(W, WT, Cout, Cin, ks * ks);
  };
  const int GBLK = (MPTS / 16 + 7) / 8;
  // g1: T1 = relu(fc0(relu(x)) + b0)   (relu-out fused; g2 reads it raw)
  auto gemm_g1 = [&](const h16* A, const h16* WT, const float* bias, h16* D) {
    gemm128_wmma_kernel<256, 1, 1, 0><<<GBLK, 256, 0, stream>>>(A, WT, bias, nullptr, D, MPTS);
  };
  // g2: T2f = fc1(T1relu) + b1 (f32 out, feeds shortcut add)
  auto gemm_g2 = [&](const h16* A, const h16* WT, const float* bias, float* D) {
    gemm128_wmma_kernel<128, 0, 0, 1><<<GBLK, 256, 0, stream>>>(A, WT, bias, nullptr, D, MPTS);
  };
  // g3: out = x @ ws + T2f
  auto gemm_g3 = [&](const h16* A, const h16* WT, const float* Cadd, h16* D) {
    gemm128_wmma_kernel<256, 0, 0, 0><<<GBLK, 256, 0, stream>>>(A, WT, nullptr, Cadd, D, MPTS);
  };
  auto conv3 = [&](const h16* X, const h16* WT, const float* bias, void* Y,
                   int Bn, int Cin, int Cout, int H, int Wd) {
    int waves = (Bn * H * Wd / 16) * (Cout / 16);
    conv_wmma_kernel<3, 1, 0><<<(waves + 7) / 8, 256, 0, stream>>>(X, WT, bias, Y,
                                                                   Bn, Cin, Cout, H, Wd);
  };
  auto conv1 = [&](const h16* X, const h16* WT, const float* bias, void* Y,
                   int Bn, int Cin, int Cout, int H, int Wd, int outF32) {
    int waves = (Bn * H * Wd / 16) * (Cout / 16);
    if (outF32)
      conv_wmma_kernel<1, 0, 1><<<(waves + 7) / 8, 256, 0, stream>>>(X, WT, bias, Y,
                                                                     Bn, Cin, Cout, H, Wd);
    else
      conv_wmma_kernel<1, 0, 0><<<(waves + 7) / 8, 256, 0, stream>>>(X, WT, bias, Y,
                                                                     Bn, Cin, Cout, H, Wd);
  };

  // ---- 0. weight conversions
  for (int k = 0; k < 5; ++k) {
    cvt_fc(blk[k].w0, w0t[k], 256, 128);
    cvt_fc(blk[k].w1, w1t[k], 128, 128);
    cvt_fc(blk[k].ws, wst[k], 256, 128);
  }
  cvt_fc(fcc_w, fcct, 128, 128);
  for (int d = 0; d < 4; ++d) {
    cvt_cv(dw[d][0], dwt[d][0], dcout[d], dcin[d][0], 3);
    cvt_cv(dw[d][1], dwt[d][1], dcout[d], dcin[d][1], 3);
  }
  for (int u = 0; u < 3; ++u) {
    cvt_cv(uw[u][0], uwt[u][0], ucout[u], 2 * ucout[u], 3);
    cvt_cv(uw[u][1], uwt[u][1], ucout[u], ucout[u], 3);
  }
  cvt_cv(fin_w, fint, 128, 32, 1);
  cvt_cv(mw, mwt, 32, 128, 1);
  cvt_cv(lw, lwt, 32, 128, 1);

  // ---- 1. plane ids + fc_pos
  ids_kernel<<<grid1d(MPTS), 256, 0, stream>>>(p_pts, IDS);
  fc_pos_kernel<<<grid1d(MPTS * 256), 256, 0, stream>>>(emb, fcpos_w, fcpos_b, ACT256h);

  // ---- 2. ResNet block 0 (no pooling)
  gemm_g1(ACT256h, w0t[0], blk[0].b0, T1h);
  gemm_g2(T1h,     w1t[0], blk[0].b1, T2f);
  gemm_g3(ACT256h, wst[0], T2f, ACT128h);

  // ---- 3. ResNet blocks 1..4 with 3-plane scatter-max pooling
  for (int k = 1; k < 5; ++k) {
    fill_u32_kernel<<<grid1d(3 * NBATCH * RSQ * HIDD), 256, 0, stream>>>(
        GMAX, 0u, 3 * NBATCH * RSQ * HIDD);
    scatter_max_kernel<<<grid1d(MPTS * HIDD), 256, 0, stream>>>(ACT128h, IDS, GMAX);
    gather_concat_kernel<<<grid1d(MPTS * HIDD), 256, 0, stream>>>(ACT128h, IDS, GMAX, ACT256h);
    gemm_g1(ACT256h, w0t[k], blk[k].b0, T1h);
    gemm_g2(T1h,     w1t[k], blk[k].b1, T2f);
    gemm_g3(ACT256h, wst[k], T2f, ACT128h);
  }

  // ---- 4. fc_c + scatter-mean -> CAT (NHWC f16 [4,192,64,128])
  gemm128_wmma_kernel<128, 0, 0, 0><<<GBLK, 256, 0, stream>>>(ACT128h, fcct, fcc_b,
                                                              nullptr, C128h, MPTS);
  fill_f32_kernel<<<grid1d(3 * NBATCH * RSQ * CDIMC), 256, 0, stream>>>(
      GSUM, 0.0f, 3 * NBATCH * RSQ * CDIMC);
  fill_f32_kernel<<<grid1d(3 * NBATCH * RSQ), 256, 0, stream>>>(GCNT, 0.0f, 3 * NBATCH * RSQ);
  scatter_sum_kernel<<<grid1d(MPTS * CDIMC), 256, 0, stream>>>(C128h, IDS, GSUM, GCNT);
  plane_cat_kernel<<<grid1d(NBATCH * 3 * RSQ * CDIMC), 256, 0, stream>>>(GSUM, GCNT, CATh);

  // ---- 5. UNet (NHWC f16)
  conv3(CATh, dwt[0][0], db[0][0], GAh, 4, 128, 32, 192, 64);
  conv3(GAh, dwt[0][1], db[0][1], SK0h, 4, 32, 32, 192, 64);
  maxpool_kernel<<<grid1d(4 * 96 * 32 * 32), 256, 0, stream>>>(SK0h, GAh, 4, 32, 192, 64);
  conv3(GAh, dwt[1][0], db[1][0], GBh, 4, 32, 64, 96, 32);
  conv3(GBh, dwt[1][1], db[1][1], SK1h, 4, 64, 64, 96, 32);
  maxpool_kernel<<<grid1d(4 * 48 * 16 * 64), 256, 0, stream>>>(SK1h, GAh, 4, 64, 96, 32);
  conv3(GAh, dwt[2][0], db[2][0], GBh, 4, 64, 128, 48, 16);
  conv3(GBh, dwt[2][1], db[2][1], SK2h, 4, 128, 128, 48, 16);
  maxpool_kernel<<<grid1d(4 * 24 * 8 * 128), 256, 0, stream>>>(SK2h, GAh, 4, 128, 48, 16);
  conv3(GAh, dwt[3][0], db[3][0], GBh, 4, 128, 256, 24, 8);
  conv3(GBh, dwt[3][1], db[3][1], GAh, 4, 256, 256, 24, 8);   // [4,24,8,256]
  // up 0: 256 -> 128 @ 48x16
  upconv_kernel<<<grid1d(4 * 48 * 16 * 128), 256, 0, stream>>>(GAh, uwu[0], ubu[0], GBh,
                                                               4, 256, 128, 24, 8, 256);
  copy_ch_kernel<<<grid1d(4 * 48 * 16 * 128), 256, 0, stream>>>(SK2h, GBh, 4 * 48 * 16, 128, 128, 256);
  conv3(GBh, uwt[0][0], ub[0][0], GAh, 4, 256, 128, 48, 16);
  conv3(GAh, uwt[0][1], ub[0][1], GBh, 4, 128, 128, 48, 16);
  // up 1: 128 -> 64 @ 96x32
  upconv_kernel<<<grid1d(4 * 96 * 32 * 64), 256, 0, stream>>>(GBh, uwu[1], ubu[1], GAh,
                                                              4, 128, 64, 48, 16, 128);
  copy_ch_kernel<<<grid1d(4 * 96 * 32 * 64), 256, 0, stream>>>(SK1h, GAh, 4 * 96 * 32, 64, 64, 128);
  conv3(GAh, uwt[1][0], ub[1][0], GBh, 4, 128, 64, 96, 32);
  conv3(GBh, uwt[1][1], ub[1][1], GAh, 4, 64, 64, 96, 32);
  // up 2: 64 -> 32 @ 192x64
  upconv_kernel<<<grid1d(4 * 192 * 64 * 32), 256, 0, stream>>>(GAh, uwu[2], ubu[2], GBh,
                                                               4, 64, 32, 96, 32, 64);
  copy_ch_kernel<<<grid1d(4 * 192 * 64 * 32), 256, 0, stream>>>(SK0h, GBh, 4 * 192 * 64, 32, 32, 64);
  conv3(GBh, uwt[2][0], ub[2][0], GAh, 4, 64, 32, 192, 64);
  conv3(GAh, uwt[2][1], ub[2][1], GBh, 4, 32, 32, 192, 64);
  // final 1x1 (no relu) -> plane_feat NHWC [4,192,64,128]
  conv1(GBh, fint, fin_b, GAh, 4, 32, 128, 192, 64, 0);

  // ---- 6. heads (f32 out) + VAE sample + KL
  conv1(GAh, mwt, mb, MEANB, 4, 128, 32, 192, 64, 1);
  conv1(GAh, lwt, lb, LOGVB, 4, 128, 32, 192, 64, 1);
  fill_f32_kernel<<<1, 256, 0, stream>>>(outp, 0.0f, 4);
  head_kernel<<<grid1d(NBATCH * LATC * 3 * RSQ), 256, 0, stream>>>(MEANB, LOGVB, noise, outp);
}
